// Codebook_1529008357391
// MI455X (gfx1250) — matrix operand order repaired and checked by reading, not compiled, not run on hardware
//
#include <hip/hip_runtime.h>
#include <hip/hip_bf16.h>

typedef __attribute__((ext_vector_type(16))) __bf16 v16bf;
typedef __attribute__((ext_vector_type(8)))  __bf16 v8bf;
typedef __attribute__((ext_vector_type(8)))  float  v8f;
typedef __attribute__((ext_vector_type(4)))  float  v4f;

#define CDIM 256
#define KCODES 8192
#define NROWS 16384
#define HWSZ 1024           // H*W per batch image
#define RPB 128             // rows of x per block (4 row-groups x 32)
#define PITCH 264           // bf16 LDS row pitch (+8 pad -> 528B, conflict-free b128)

// ---------------- kernel 1: split codebook into bf16 hi/lo + row norms ----
__global__ __launch_bounds__(256)
void prep_e_kernel(const float* __restrict__ e,
                   __bf16* __restrict__ ehi,
                   __bf16* __restrict__ elo,
                   float* __restrict__ enorm) {
    __shared__ float red[8];
    const int k = blockIdx.x;
    const int t = threadIdx.x;            // t == c
    float v = e[(size_t)k * CDIM + t];
    __bf16 hi = (__bf16)v;
    float lo = v - (float)hi;
    ehi[(size_t)k * CDIM + t] = hi;
    elo[(size_t)k * CDIM + t] = (__bf16)lo;
    float s = v * v;
    #pragma unroll
    for (int m = 16; m >= 1; m >>= 1) s += __shfl_xor(s, m, 32);
    if ((t & 31) == 0) red[t >> 5] = s;
    __syncthreads();
    if (t == 0) {
        float tot = 0.f;
        #pragma unroll
        for (int w = 0; w < 8; ++w) tot += red[w];
        enorm[k] = tot;
    }
}

// ---------------- kernel 2: fused GEMM + argmin + gather + transposes -----
__global__ __launch_bounds__(256)
void vq_kernel(const float* __restrict__ x,
               const float* __restrict__ e,
               const __bf16* __restrict__ ehi,
               const __bf16* __restrict__ elo,
               const float* __restrict__ enorm,
               float* __restrict__ out_q_bchw,   // x_q_detach == x_q, BCHW
               float* __restrict__ out_q_bhwc,   // x_q, BHWC
               float* __restrict__ out_fea) {    // x_fea, BHWC
    __shared__ __align__(16) __bf16 xhi[RPB][PITCH];
    __shared__ __align__(16) __bf16 xlo[RPB][PITCH];
    __shared__ float wbv[2][RPB];
    __shared__ int   wbi[2][RPB];
    __shared__ int   fidx[RPB];

    const int t   = threadIdx.x;          // phase A/D: t == channel c
    const int n0  = blockIdx.x * RPB;     // first flat row (b*1024 + h*32 + w)
    const int b   = n0 / HWSZ;            // 128 | 1024: one batch per block
    const int hw0 = n0 % HWSZ;

    // ---- phase A: stage x rows as bf16 hi/lo in LDS, emit x_fea ----------
    const float* xs = x + (size_t)b * CDIM * HWSZ + (size_t)t * HWSZ + hw0;
    for (int r4 = 0; r4 < RPB; r4 += 4) {
        v4f v = *(const v4f*)(xs + r4);
        #pragma unroll
        for (int i = 0; i < 4; ++i) {
            const int r = r4 + i;
            const float f = v[i];
            const __bf16 hi = (__bf16)f;
            xhi[r][t] = hi;
            xlo[r][t] = (__bf16)(f - (float)hi);
            out_fea[(size_t)(n0 + r) * CDIM + t] = f;
        }
    }
    __syncthreads();

    // ---- wave mapping: 4 row-groups x 2 code-strips ----------------------
    const int wave   = t >> 5;
    const int rg     = wave >> 1;         // row group 0..3 (32 rows each)
    const int cs     = wave & 1;          // code strip 0..1
    const int lane   = t & 31;
    const int nIdx   = lane & 15;         // code column within 16x16 tile
    const int hiHalf = lane >> 4;         // 0: lanes 0-15, 1: lanes 16-31

    // ---- load A fragments ONCE into registers (explicit hoist) -----------
    // 8 c-steps x 2 m-tiles x {hi,lo} x v16bf = 256 VGPRs, loop-invariant.
    v16bf Ah[8][2], Al[8][2];
    #pragma unroll
    for (int s8 = 0; s8 < 8; ++s8) {
        const int c0 = s8 * 32;
        #pragma unroll
        for (int m = 0; m < 2; ++m) {
            const int row = rg * 32 + m * 16 + nIdx;  // A lanes 0-15/16-31: M=0-15
            const __bf16* pah = &xhi[row][c0 + hiHalf * 16];
            const __bf16* pal = &xlo[row][c0 + hiHalf * 16];
            const v8bf a0 = *(const v8bf*)(pah);
            const v8bf a1 = *(const v8bf*)(pah + 8);
            const v8bf l0 = *(const v8bf*)(pal);
            const v8bf l1 = *(const v8bf*)(pal + 8);
            Ah[s8][m] = __builtin_shufflevector(a0, a1,
                0,1,2,3,4,5,6,7,8,9,10,11,12,13,14,15);
            Al[s8][m] = __builtin_shufflevector(l0, l1,
                0,1,2,3,4,5,6,7,8,9,10,11,12,13,14,15);
        }
    }

    // ---- phase B: sweep codes; per-lane running argmin -------------------
    const v8f azero = {};
    float bv[2][8];
    int   bi[2][8];
    #pragma unroll
    for (int m = 0; m < 2; ++m)
        #pragma unroll
        for (int j = 0; j < 8; ++j) { bv[m][j] = 3.4e38f; bi[m][j] = 0; }

    for (int it = 0; it < KCODES / 32; ++it) {
        const int code = it * 32 + cs * 16 + nIdx;
        const float en = enorm[code];
        const __bf16* pbh = ehi + (size_t)code * CDIM + hiHalf * 16;
        const __bf16* pbl = elo + (size_t)code * CDIM + hiHalf * 16;
        const int pf = (it + 1 < KCODES / 32) ? 32 * CDIM : 0;
        __builtin_prefetch(pbh + pf, 0, 1);
        __builtin_prefetch(pbl + pf, 0, 1);

        // two partial accumulators per m-tile -> 4 independent WMMA chains
        v8f acc0a = azero, acc0b = azero, acc1a = azero, acc1b = azero;
        #pragma unroll
        for (int s8 = 0; s8 < 8; s8 += 2) {
            const v16bf bh0 = *(const v16bf*)(pbh + s8 * 32);
            const v16bf bl0 = *(const v16bf*)(pbl + s8 * 32);
            const v16bf bh1 = *(const v16bf*)(pbh + s8 * 32 + 32);
            const v16bf bl1 = *(const v16bf*)(pbl + s8 * 32 + 32);
            // x.e ~= xhi.ehi + xhi.elo + xlo.ehi  (f32 accumulate)
            acc0a = __builtin_amdgcn_wmma_f32_16x16x32_bf16(false, Ah[s8][0],   false, bh0, (short)0, acc0a, false, false);
            acc1a = __builtin_amdgcn_wmma_f32_16x16x32_bf16(false, Ah[s8][1],   false, bh0, (short)0, acc1a, false, false);
            acc0b = __builtin_amdgcn_wmma_f32_16x16x32_bf16(false, Ah[s8+1][0], false, bh1, (short)0, acc0b, false, false);
            acc1b = __builtin_amdgcn_wmma_f32_16x16x32_bf16(false, Ah[s8+1][1], false, bh1, (short)0, acc1b, false, false);
            acc0a = __builtin_amdgcn_wmma_f32_16x16x32_bf16(false, Ah[s8][0],   false, bl0, (short)0, acc0a, false, false);
            acc1a = __builtin_amdgcn_wmma_f32_16x16x32_bf16(false, Ah[s8][1],   false, bl0, (short)0, acc1a, false, false);
            acc0b = __builtin_amdgcn_wmma_f32_16x16x32_bf16(false, Ah[s8+1][0], false, bl1, (short)0, acc0b, false, false);
            acc1b = __builtin_amdgcn_wmma_f32_16x16x32_bf16(false, Ah[s8+1][1], false, bl1, (short)0, acc1b, false, false);
            acc0a = __builtin_amdgcn_wmma_f32_16x16x32_bf16(false, Al[s8][0],   false, bh0, (short)0, acc0a, false, false);
            acc1a = __builtin_amdgcn_wmma_f32_16x16x32_bf16(false, Al[s8][1],   false, bh0, (short)0, acc1a, false, false);
            acc0b = __builtin_amdgcn_wmma_f32_16x16x32_bf16(false, Al[s8+1][0], false, bh1, (short)0, acc0b, false, false);
            acc1b = __builtin_amdgcn_wmma_f32_16x16x32_bf16(false, Al[s8+1][1], false, bh1, (short)0, acc1b, false, false);
        }
        const v8f acc0 = acc0a + acc0b;
        const v8f acc1 = acc1a + acc1b;

        // distance proxy: ||e||^2 - 2 x.e  (||x||^2 constant per row)
        #pragma unroll
        for (int j = 0; j < 8; ++j) {
            const float d0 = en - 2.0f * acc0[j];
            if (d0 < bv[0][j]) { bv[0][j] = d0; bi[0][j] = code; }
            const float d1 = en - 2.0f * acc1[j];
            if (d1 < bv[1][j]) { bv[1][j] = d1; bi[1][j] = code; }
        }
    }

    // ---- phase C: argmin reduction (16 lanes -> LDS -> strip pair) -------
    #pragma unroll
    for (int m = 0; m < 2; ++m)
        #pragma unroll
        for (int j = 0; j < 8; ++j) {
            float v = bv[m][j];
            int   idx = bi[m][j];
            #pragma unroll
            for (int s = 1; s < 16; s <<= 1) {
                const float ov = __shfl_xor(v, s, 32);
                const int   oi = __shfl_xor(idx, s, 32);
                if (ov < v || (ov == v && oi < idx)) { v = ov; idx = oi; }
            }
            if (nIdx == 0) {                       // D layout: lanes16-31 = M+8
                const int row = rg * 32 + m * 16 + hiHalf * 8 + j;
                wbv[cs][row] = v;
                wbi[cs][row] = idx;
            }
        }
    __syncthreads();
    if (t < RPB) {
        float v = wbv[0][t];
        int idx = wbi[0][t];
        const float ov = wbv[1][t];
        const int   oi = wbi[1][t];
        if (ov < v || (ov == v && oi < idx)) { v = ov; idx = oi; }
        fidx[t] = idx;
    }
    __syncthreads();

    // ---- phase D: gather exact f32 codebook rows, write both layouts -----
    for (int r4 = 0; r4 < RPB; r4 += 4) {
        v4f o;
        #pragma unroll
        for (int i = 0; i < 4; ++i) {
            const int r = r4 + i;
            const float val = e[(size_t)fidx[r] * CDIM + t];
            o[i] = val;
            out_q_bhwc[(size_t)(n0 + r) * CDIM + t] = val;   // coalesced over t
        }
        *(v4f*)(out_q_bchw + (size_t)b * CDIM * HWSZ + (size_t)t * HWSZ
                + hw0 + r4) = o;                              // contiguous over r
    }
}

extern "C" void kernel_launch(void* const* d_in, const int* in_sizes, int n_in,
                              void* d_out, int out_size, void* d_ws, size_t ws_size,
                              hipStream_t stream) {
    const float* x = (const float*)d_in[0];            // (16,256,32,32) f32
    const float* e = (const float*)d_in[1];            // (8192,256) f32

    char* ws = (char*)d_ws;                            // needs ~8.42 MB
    __bf16* ehi  = (__bf16*)ws;
    __bf16* elo  = (__bf16*)(ws + (size_t)KCODES * CDIM * sizeof(__bf16));
    float* enorm = (float*)(ws + (size_t)KCODES * CDIM * 2 * sizeof(__bf16));

    float* out0 = (float*)d_out;                       // x_q_detach (B,C,H,W)
    float* out1 = out0 + (size_t)NROWS * CDIM;         // x_q        (B,H,W,C)
    float* out2 = out1 + (size_t)NROWS * CDIM;         // x_fea      (B,H,W,C)

    prep_e_kernel<<<KCODES, 256, 0, stream>>>(e, ehi, elo, enorm);
    vq_kernel<<<NROWS / RPB, 256, 0, stream>>>(x, e, ehi, elo, enorm,
                                               out0, out1, out2);
}
